// SimilarityTreeLSTM_68676527063709
// MI455X (gfx1250) — matrix operand order repaired and checked by reading, not compiled, not run on hardware
//
#include <hip/hip_runtime.h>
#include <hip/hip_bf16.h>
#include <stdint.h>
#include <stddef.h>

#define N_NODES 256
#define IN_DIM  300
#define KP      320   // padded K for X GEMMs (mult of 32)
#define M_DIM   150
#define M3      450
#define M3P     456   // padded width of transposed iouh_w rows
#define MW      152   // padded width of transposed fh_w / attnh_w-left rows
#define OCAT    608   // padded 600 = 450 (iou) + 150 (fx)
#define MP      160   // padded M (mult of 32)
#define HID_DIM 50
#define NCLS    5
#define SCAN_BD 512   // scan block: 16 waves on one WGP

typedef __attribute__((ext_vector_type(16))) __bf16 v16bf;
typedef __attribute__((ext_vector_type(8)))  float  v8f;

__device__ __forceinline__ float sigmoidf_(float x) { return 1.0f / (1.0f + __expf(-x)); }

// ---------------------------------------------------------------------------
// Conversion / gather kernels (build padded bf16 operands once per call)
// ---------------------------------------------------------------------------
__global__ void build_x_bf16(const float* __restrict__ emb, const int* __restrict__ toks,
                             __bf16* __restrict__ X) {
  int i = blockIdx.x;        // node
  int k = threadIdx.x;       // 0..KP-1 (blockDim = KP = 320)
  float v = (k < IN_DIM) ? emb[(size_t)toks[i] * IN_DIM + k] : 0.0f;
  X[(size_t)i * KP + k] = (__bf16)v;
}

__global__ void build_wcat_bf16(const float* __restrict__ ioux_w, const float* __restrict__ fx_w,
                                const float* __restrict__ ioux_b, const float* __restrict__ fx_b,
                                const float* __restrict__ iouh_b,
                                __bf16* __restrict__ Wcat, float* __restrict__ biascat) {
  int o = blockIdx.x;        // 0..OCAT-1
  int k = threadIdx.x;       // 0..KP-1
  float v = 0.0f;
  if (k < IN_DIM) {
    if (o < M3)              v = ioux_w[(size_t)o * IN_DIM + k];
    else if (o < M3 + M_DIM) v = fx_w[(size_t)(o - M3) * IN_DIM + k];
  }
  Wcat[(size_t)o * KP + k] = (__bf16)v;
  // fold ioux_b + iouh_b (both per-column constants in every step) into GEMM bias
  if (k == 0) biascat[o] = (o < M3) ? (ioux_b[o] + iouh_b[o])
                                    : ((o < M3 + M_DIM) ? fx_b[o - M3] : 0.0f);
}

__global__ void build_wr_bf16(const float* __restrict__ attnh_w, __bf16* __restrict__ Wr) {
  int o = blockIdx.x;        // 0..MP-1
  int k = threadIdx.x;       // 0..MP-1
  float v = (o < M_DIM && k < M_DIM) ? attnh_w[(size_t)o * (2 * M_DIM) + M_DIM + k] : 0.0f;
  Wr[(size_t)o * MP + k] = (__bf16)v;
}

__global__ void h_to_bf16(const float* __restrict__ H, __bf16* __restrict__ Hb) {
  int i = blockIdx.x;        // node
  int k = threadIdx.x;       // 0..MP-1
  Hb[(size_t)i * MP + k] = (__bf16)((k < M_DIM) ? H[(size_t)i * M_DIM + k] : 0.0f);
}

// Transposed f32 weight copies so the scan's matvec inner loops are
// unit-stride across lanes (full cacheline utilization from WGP$/L2).
__global__ void build_iouhT(const float* __restrict__ iouh_w, float* __restrict__ iouh_wT) {
  int k = blockIdx.x;        // 0..M_DIM-1
  int o = threadIdx.x;       // blockDim = 512, guard
  if (o < M3P)
    iouh_wT[(size_t)k * M3P + o] = (o < M3) ? iouh_w[(size_t)o * M_DIM + k] : 0.0f;
}

__global__ void build_fhT_attnLT(const float* __restrict__ fh_w, const float* __restrict__ attnh_w,
                                 float* __restrict__ fh_wT, float* __restrict__ attnhL_T) {
  int k = blockIdx.x;        // 0..M_DIM-1
  int j = threadIdx.x;       // blockDim = MW*? use 160, guard
  if (j < MW) {
    fh_wT[(size_t)k * MW + j]    = (j < M_DIM) ? fh_w[(size_t)j * M_DIM + k] : 0.0f;
    attnhL_T[(size_t)k * MW + j] = (j < M_DIM) ? attnh_w[(size_t)j * (2 * M_DIM) + k] : 0.0f;
  }
}

// ---------------------------------------------------------------------------
// WMMA bf16 GEMM: C[m][n] = sum_k A[m][k] * W[n][k]  (+ bias[n])
// One wave per 16x16 output tile. Lane layouts per CDNA5 ISA 7.12.2.
// ---------------------------------------------------------------------------
template<int K, bool TSTORE>
__global__ void __launch_bounds__(32)
wmma_gemm_bf16(const __bf16* __restrict__ A, int lda,
               const __bf16* __restrict__ W, int ldw,
               const float* __restrict__ bias,
               float* __restrict__ C, int ldc)
{
  const int lane = threadIdx.x & 31;
  const int r    = lane & 15;
  const int hi   = lane >> 4;
  const int n0   = blockIdx.x * 16;
  const int m0   = blockIdx.y * 16;
  const __bf16* arow = A + (size_t)(m0 + r) * lda;   // A: lane r holds row m0+r
  const __bf16* wrow = W + (size_t)(n0 + r) * ldw;   // B[k][n] = W[n0+n][k]
  v8f acc = {0.f, 0.f, 0.f, 0.f, 0.f, 0.f, 0.f, 0.f};
#pragma unroll
  for (int k0 = 0; k0 < K; k0 += 32) {
    v16bf a, b;
#pragma unroll
    for (int j = 0; j < 8; ++j) {
      // A 16x32 bf16: VGPR j<4 -> K = hi*8 + 2j {,+1}; j>=4 -> 16 + hi*8 + 2(j-4)
      const int ka = (j < 4 ? 0 : 16) + hi * 8 + 2 * (j & 3);
      a[2 * j]     = arow[k0 + ka];
      a[2 * j + 1] = arow[k0 + ka + 1];
      // B 32x16 bf16: same K layout as A, lane r = column
      const int kb = (j < 4 ? 0 : 16) + hi * 8 + 2 * (j & 3);
      b[2 * j]     = wrow[k0 + kb];
      b[2 * j + 1] = wrow[k0 + kb + 1];
    }
    acc = __builtin_amdgcn_wmma_f32_16x16x32_bf16(false, a, false, b, (short)0, acc, false, false);
  }
  // D 16x16 f32: acc[j] = D[hi*8 + j][r]
  if (TSTORE) {
#pragma unroll
    for (int j = 0; j < 8; ++j)
      C[(size_t)(n0 + r) * ldc + (m0 + hi * 8 + j)] = acc[j];
  } else {
    const float bv = bias ? bias[n0 + r] : 0.0f;
#pragma unroll
    for (int j = 0; j < 8; ++j)
      C[(size_t)(m0 + hi * 8 + j) * ldc + (n0 + r)] = acc[j] + bv;
  }
}

// ---------------------------------------------------------------------------
// Sequential tree scan (one block per tree; 512 threads). Optional attention.
// Ccat[i][0..449]  = x@ioux_w^T + ioux_b + iouh_b   (iou_x with both biases)
// Ccat[i][450..599]= x@fx_w^T   + fx_b              (fx_all rows; row 256 == 0)
// ApreT[j][l]      = Hattn[l] . attnh_w[j,150:]     (transposed for coalescing)
// ---------------------------------------------------------------------------
struct ScanArgs {
  const float* Ccat;
  const int*   par;
  const float* iouh_wT;           // [M_DIM][M3P]
  const float* fh_wT;             // [M_DIM][MW]
  const float* fh_b;
  const float* attnhL_T;          // [M_DIM][MW]
  const float* attnh_b; const float* Wa;
  const float* ApreT;             // [MP][N_NODES] or nullptr
  const float* Hattn;             // [N_NODES][M_DIM] or nullptr
  float* h_sum; float* fc_sum;    // [ (N_NODES+1) * M_DIM ] each
  float* h_all;                   // [ N_NODES * M_DIM ]
};

__global__ void __launch_bounds__(SCAN_BD) tree_scan(ScanArgs a0, ScanArgs a1)
{
  ScanArgs A = (blockIdx.x == 0) ? a0 : a1;
  const int tid = threadIdx.x;

  __shared__ float sh_h[M_DIM], sh_c[M_DIM], sh_hs[M_DIM], sh_ha[M_DIM];
  __shared__ float sh_iou[M3];
  __shared__ float sh_s[N_NODES], sh_r[N_NODES];
  __shared__ float sh_wa[M_DIM], sh_ab[M_DIM], sh_fb[M_DIM];

  // zero accumulators (deterministic across graph replays) + stage constants
  for (int idx = tid; idx < (N_NODES + 1) * M_DIM; idx += SCAN_BD) {
    A.h_sum[idx] = 0.0f; A.fc_sum[idx] = 0.0f;
  }
  for (int j = tid; j < M_DIM; j += SCAN_BD) {
    sh_wa[j] = A.Wa[j]; sh_ab[j] = A.attnh_b[j]; sh_fb[j] = A.fh_b[j];
  }
  __syncthreads();

  for (int i = 0; i < N_NODES; ++i) {
    if (tid < M_DIM) sh_hs[tid] = A.h_sum[i * M_DIM + tid];   // same lane wrote it earlier
    __syncthreads();

    // iou = iou_x[i] + h_sum[i] @ iouh_w^T (+ biases already folded into Ccat)
    if (tid < M3) {
      float acc = A.Ccat[(size_t)i * OCAT + tid];
      for (int k = 0; k < M_DIM; ++k)
        acc = fmaf(A.iouh_wT[(size_t)k * M3P + tid], sh_hs[k], acc);  // lane-coalesced
      sh_iou[tid] = acc;
    }
    __syncthreads();

    if (tid < M_DIM) {
      float ig = sh_iou[tid], og = sh_iou[M_DIM + tid], ug = sh_iou[2 * M_DIM + tid];
      float c = sigmoidf_(ig) * tanhf(ug) + A.fc_sum[i * M_DIM + tid];
      sh_c[tid] = c;
      sh_h[tid] = sigmoidf_(og) * tanhf(c);
    }
    __syncthreads();

    if (A.Hattn != nullptr) {
      // ha = h @ attnh_w[:, :150]^T + attnh_b   (lane-coalesced transposed reads)
      if (tid < M_DIM) {
        float acc = sh_ab[tid];
        for (int k = 0; k < M_DIM; ++k)
          acc = fmaf(A.attnhL_T[(size_t)k * MW + tid], sh_h[k], acc);
        sh_ha[tid] = acc;
      }
      __syncthreads();
      // raw scores: s_l = tanh(ha + Apre[l]) . Wa   (one l per thread, coalesced ApreT)
      float e = 0.0f;
      if (tid < N_NODES) {
        float acc = 0.0f;
        for (int j = 0; j < M_DIM; ++j) {
          float t = tanhf(sh_ha[j] + A.ApreT[(size_t)j * N_NODES + tid]);
          acc = fmaf(t, sh_wa[j], acc);
        }
        sh_s[tid] = acc;
        sh_r[tid] = acc;
      }
      __syncthreads();
      // softmax over 256
      for (int st = 128; st > 0; st >>= 1) {
        if (tid < st) sh_r[tid] = fmaxf(sh_r[tid], sh_r[tid + st]);
        __syncthreads();
      }
      float mx = sh_r[0];
      __syncthreads();
      if (tid < N_NODES) { e = __expf(sh_s[tid] - mx); sh_r[tid] = e; }
      __syncthreads();
      for (int st = 128; st > 0; st >>= 1) {
        if (tid < st) sh_r[tid] += sh_r[tid + st];
        __syncthreads();
      }
      float denom = sh_r[0];
      __syncthreads();
      if (tid < N_NODES) sh_s[tid] = e / denom;
      __syncthreads();
      // h = sum_l (1 - s_l) * Hattn[l] + (sum s) * h;  sum(softmax) == 1
      if (tid < M_DIM) {
        float acc = 0.0f;
        for (int l = 0; l < N_NODES; ++l)
          acc = fmaf(1.0f - sh_s[l], A.Hattn[(size_t)l * M_DIM + tid], acc);
        sh_ha[tid] = acc + sh_h[tid];
      }
      __syncthreads();
      if (tid < M_DIM) sh_h[tid] = sh_ha[tid];
      __syncthreads();
    }

    // f = sigmoid(h @ fh_w^T + fh_b + fx_all[p]); scatter-add to parent
    const int p = A.par[i];
    if (tid < M_DIM) {
      float acc = sh_fb[tid];
      for (int k = 0; k < M_DIM; ++k)
        acc = fmaf(A.fh_wT[(size_t)k * MW + tid], sh_h[k], acc);      // lane-coalesced
      float fx = (p < N_NODES) ? A.Ccat[(size_t)p * OCAT + M3 + tid] : 0.0f;
      float f = sigmoidf_(acc + fx);
      A.h_sum[p * M_DIM + tid]  += sh_h[tid];      // same lane re-reads later: program order
      A.fc_sum[p * M_DIM + tid] += f * sh_c[tid];
      A.h_all[(size_t)i * M_DIM + tid] = sh_h[tid];
    }
    __syncthreads();
  }
}

// ---------------------------------------------------------------------------
// Final classifier: lhid/rhid -> sigmoid MLP -> log_softmax -> out[5]
// ---------------------------------------------------------------------------
__global__ void __launch_bounds__(64)
final_head(const float* __restrict__ Hl_all, const float* __restrict__ Hr_all,
           const float* __restrict__ Hrh1,  const float* __restrict__ Hlh2,
           const float* __restrict__ wh_w, const float* __restrict__ wh_b,
           const float* __restrict__ wp_w, const float* __restrict__ wp_b,
           float* __restrict__ out)
{
  __shared__ float vec[2 * M_DIM], hid[HID_DIM], logit[NCLS];
  const int tid = threadIdx.x;
  const size_t root = (size_t)(N_NODES - 1) * M_DIM;
  for (int j = tid; j < M_DIM; j += 64) {
    float lh = tanhf(Hl_all[root + j] + Hlh2[root + j]);
    float rh = tanhf(Hrh1[root + j] + Hr_all[root + j]);
    vec[j]         = lh * rh;
    vec[M_DIM + j] = fabsf(lh - rh);
  }
  __syncthreads();
  if (tid < HID_DIM) {
    const float* wr = wh_w + (size_t)tid * (2 * M_DIM);
    float acc = wh_b[tid];
    for (int k = 0; k < 2 * M_DIM; ++k) acc = fmaf(wr[k], vec[k], acc);
    hid[tid] = sigmoidf_(acc);
  }
  __syncthreads();
  if (tid < NCLS) {
    const float* wr = wp_w + (size_t)tid * HID_DIM;
    float acc = wp_b[tid];
    for (int k = 0; k < HID_DIM; ++k) acc = fmaf(wr[k], hid[k], acc);
    logit[tid] = acc;
  }
  __syncthreads();
  if (tid == 0) {
    float mx = logit[0];
    for (int j = 1; j < NCLS; ++j) mx = fmaxf(mx, logit[j]);
    float s = 0.0f;
    for (int j = 0; j < NCLS; ++j) s += __expf(logit[j] - mx);
    float lse = mx + __logf(s);
    for (int j = 0; j < NCLS; ++j) out[j] = logit[j] - lse;
  }
}

// ---------------------------------------------------------------------------
extern "C" void kernel_launch(void* const* d_in, const int* in_sizes, int n_in,
                              void* d_out, int out_size, void* d_ws, size_t ws_size,
                              hipStream_t stream)
{
  (void)in_sizes; (void)n_in; (void)out_size; (void)ws_size;
  const float* emb     = (const float*)d_in[0];
  const float* ioux_w  = (const float*)d_in[1];
  const float* ioux_b  = (const float*)d_in[2];
  const float* iouh_w  = (const float*)d_in[3];
  const float* iouh_b  = (const float*)d_in[4];
  const float* fx_w    = (const float*)d_in[5];
  const float* fx_b    = (const float*)d_in[6];
  const float* fh_w    = (const float*)d_in[7];
  const float* fh_b    = (const float*)d_in[8];
  const float* Wa      = (const float*)d_in[9];
  const float* attnh_w = (const float*)d_in[10];
  const float* attnh_b = (const float*)d_in[11];
  const float* wh_w    = (const float*)d_in[12];
  const float* wh_b    = (const float*)d_in[13];
  const float* wp_w    = (const float*)d_in[14];
  const float* wp_b    = (const float*)d_in[15];
  const int* ltoks = (const int*)d_in[16];
  const int* rtoks = (const int*)d_in[17];
  const int* lpar  = (const int*)d_in[18];
  const int* rpar  = (const int*)d_in[19];
  float* out = (float*)d_out;

  // bump allocator over d_ws
  uintptr_t cur = (uintptr_t)d_ws;
  auto alloc = [&](size_t bytes) -> void* {
    uintptr_t p = (cur + 255) & ~(uintptr_t)255;
    cur = p + bytes;
    return (void*)p;
  };
  __bf16* Xl       = (__bf16*)alloc((size_t)N_NODES * KP * sizeof(__bf16));
  __bf16* Xr       = (__bf16*)alloc((size_t)N_NODES * KP * sizeof(__bf16));
  __bf16* Wcat     = (__bf16*)alloc((size_t)OCAT * KP * sizeof(__bf16));
  float*  biascat  = (float*) alloc((size_t)OCAT * sizeof(float));
  __bf16* Wr       = (__bf16*)alloc((size_t)MP * MP * sizeof(__bf16));
  float*  iouh_wT  = (float*) alloc((size_t)M_DIM * M3P * sizeof(float));
  float*  fh_wT    = (float*) alloc((size_t)M_DIM * MW * sizeof(float));
  float*  attnhL_T = (float*) alloc((size_t)M_DIM * MW * sizeof(float));
  float*  Ccat_l   = (float*) alloc((size_t)N_NODES * OCAT * sizeof(float));
  float*  Ccat_r   = (float*) alloc((size_t)N_NODES * OCAT * sizeof(float));
  float*  hsumA    = (float*) alloc((size_t)(N_NODES + 1) * M_DIM * sizeof(float));
  float*  fcsumA   = (float*) alloc((size_t)(N_NODES + 1) * M_DIM * sizeof(float));
  float*  hsumB    = (float*) alloc((size_t)(N_NODES + 1) * M_DIM * sizeof(float));
  float*  fcsumB   = (float*) alloc((size_t)(N_NODES + 1) * M_DIM * sizeof(float));
  float*  Hl_all   = (float*) alloc((size_t)N_NODES * M_DIM * sizeof(float));
  float*  Hr_all   = (float*) alloc((size_t)N_NODES * M_DIM * sizeof(float));
  float*  Hrh1     = (float*) alloc((size_t)N_NODES * M_DIM * sizeof(float));
  float*  Hlh2     = (float*) alloc((size_t)N_NODES * M_DIM * sizeof(float));
  __bf16* Hlb      = (__bf16*)alloc((size_t)N_NODES * MP * sizeof(__bf16));
  __bf16* Hrb      = (__bf16*)alloc((size_t)N_NODES * MP * sizeof(__bf16));
  float*  ApreT_l  = (float*) alloc((size_t)MP * N_NODES * sizeof(float));
  float*  ApreT_r  = (float*) alloc((size_t)MP * N_NODES * sizeof(float));

  // 1) build padded bf16 operands + transposed f32 weights for the scan
  build_x_bf16<<<N_NODES, KP, 0, stream>>>(emb, ltoks, Xl);
  build_x_bf16<<<N_NODES, KP, 0, stream>>>(emb, rtoks, Xr);
  build_wcat_bf16<<<OCAT, KP, 0, stream>>>(ioux_w, fx_w, ioux_b, fx_b, iouh_b, Wcat, biascat);
  build_wr_bf16<<<MP, MP, 0, stream>>>(attnh_w, Wr);
  build_iouhT<<<M_DIM, 512, 0, stream>>>(iouh_w, iouh_wT);
  build_fhT_attnLT<<<M_DIM, 160, 0, stream>>>(fh_w, attnh_w, fh_wT, attnhL_T);

  // 2) WMMA GEMMs: Ccat = X @ Wcat^T + biascat   (256 x 608, K=320)
  {
    dim3 g(OCAT / 16, N_NODES / 16);
    wmma_gemm_bf16<KP, false><<<g, 32, 0, stream>>>(Xl, KP, Wcat, KP, biascat, Ccat_l, OCAT);
    wmma_gemm_bf16<KP, false><<<g, 32, 0, stream>>>(Xr, KP, Wcat, KP, biascat, Ccat_r, OCAT);
  }

  // 3) plain passes (both trees in parallel, one block each)
  {
    ScanArgs aL = { Ccat_l, lpar, iouh_wT, fh_wT, fh_b, attnhL_T, attnh_b, Wa,
                    nullptr, nullptr, hsumA, fcsumA, Hl_all };
    ScanArgs aR = { Ccat_r, rpar, iouh_wT, fh_wT, fh_b, attnhL_T, attnh_b, Wa,
                    nullptr, nullptr, hsumB, fcsumB, Hr_all };
    tree_scan<<<2, SCAN_BD, 0, stream>>>(aL, aR);
  }

  // 4) attention precompute: ApreT = (H @ attnh_w[:,150:]^T)^T   (K=160)
  h_to_bf16<<<N_NODES, MP, 0, stream>>>(Hl_all, Hlb);
  h_to_bf16<<<N_NODES, MP, 0, stream>>>(Hr_all, Hrb);
  {
    dim3 g(MP / 16, N_NODES / 16);
    wmma_gemm_bf16<MP, true><<<g, 32, 0, stream>>>(Hlb, MP, Wr, MP, nullptr, ApreT_l, N_NODES);
    wmma_gemm_bf16<MP, true><<<g, 32, 0, stream>>>(Hrb, MP, Wr, MP, nullptr, ApreT_r, N_NODES);
  }

  // 5) attention passes: rh1 = pass(right tree, attn over Hl); lh2 = pass(left tree, attn over Hr)
  {
    ScanArgs aRH1 = { Ccat_r, rpar, iouh_wT, fh_wT, fh_b, attnhL_T, attnh_b, Wa,
                      ApreT_l, Hl_all, hsumA, fcsumA, Hrh1 };
    ScanArgs aLH2 = { Ccat_l, lpar, iouh_wT, fh_wT, fh_b, attnhL_T, attnh_b, Wa,
                      ApreT_r, Hr_all, hsumB, fcsumB, Hlh2 };
    tree_scan<<<2, SCAN_BD, 0, stream>>>(aRH1, aLH2);
  }

  // 6) classifier head
  final_head<<<1, 64, 0, stream>>>(Hl_all, Hr_all, Hrh1, Hlh2, wh_w, wh_b, wp_w, wp_b, out);
}
